// LinearAttention_59828894433775
// MI455X (gfx1250) — matrix operand order repaired and checked by reading, compiled, not run on hardware
//
#include <hip/hip_runtime.h>
#include <hip/hip_bf16.h>

typedef __bf16 bf16_t;
typedef __attribute__((ext_vector_type(16))) __bf16 v16bf;
typedef __attribute__((ext_vector_type(8)))  __bf16 v8bf;
typedef __attribute__((ext_vector_type(4)))  __bf16 v4bf;
typedef __attribute__((ext_vector_type(8)))  float  v8f;

union ABFrag { v16bf v; v8bf h[2]; };

// One double-bufferable set of operand fragments for a 32x64 wave tile,
// one 32-wide K step: 2 A fragments (16x32 each) + 4 B fragments (32x16 each).
struct Frags { ABFrag a0, a1, b[4]; };

__device__ __forceinline__ void load_frags(Frags& f,
                                           const bf16_t* __restrict__ arow0,
                                           const bf16_t* __restrict__ arow1,
                                           const bf16_t* __restrict__ brow,
                                           int ldb, int k, int half) {
  f.a0.h[0] = *(const v8bf*)(arow0 + k + 8 * half);
  f.a0.h[1] = *(const v8bf*)(arow0 + k + 16 + 8 * half);
  f.a1.h[0] = *(const v8bf*)(arow1 + k + 8 * half);
  f.a1.h[1] = *(const v8bf*)(arow1 + k + 16 + 8 * half);
#pragma unroll
  for (int j = 0; j < 4; ++j) {
    const bf16_t* br = brow + (size_t)(16 * j) * ldb + k + 8 * half;
    f.b[j].h[0] = *(const v8bf*)(br);
    f.b[j].h[1] = *(const v8bf*)(br + 16);
  }
}

__device__ __forceinline__ void do_mma(v8f acc[2][4], const Frags& f) {
#pragma unroll
  for (int j = 0; j < 4; ++j) {
    acc[0][j] = __builtin_amdgcn_wmma_f32_16x16x32_bf16(
        false, f.a0.v, false, f.b[j].v, (short)0, acc[0][j], false, false);
    acc[1][j] = __builtin_amdgcn_wmma_f32_16x16x32_bf16(
        false, f.a1.v, false, f.b[j].v, (short)0, acc[1][j], false, false);
  }
}

// ---------------------------------------------------------------------------
// f32 -> bf16 elementwise convert (vectorized x4)
// ---------------------------------------------------------------------------
__global__ void k_f32_to_bf16x4(const float* __restrict__ in,
                                bf16_t* __restrict__ out) {
  const int i = (blockIdx.x * blockDim.x + threadIdx.x) * 4;
  float4 f = *(const float4*)(in + i);
  v4bf o;
  o[0] = (bf16_t)f.x; o[1] = (bf16_t)f.y; o[2] = (bf16_t)f.z; o[3] = (bf16_t)f.w;
  *(v4bf*)(out + i) = o;
}

// ---------------------------------------------------------------------------
// W [rows,cols] f32 (row-major) -> Wt [cols,rows] bf16 (row-major, K-major)
// ---------------------------------------------------------------------------
__global__ void k_transpose_to_bf16(const float* __restrict__ w,
                                    bf16_t* __restrict__ wt,
                                    int rows, int cols) {
  const int idx = blockIdx.x * blockDim.x + threadIdx.x;  // over rows*cols
  const int k = idx / cols;
  const int n = idx % cols;
  wt[(size_t)n * rows + k] = (bf16_t)w[idx];
}

// ---------------------------------------------------------------------------
// Batched bf16 GEMM: C = alpha * A @ Bt^T + bias
//   A  : [M, Kdim]  row-major bf16 (per batch, stride strideA)
//   Bt : [N, Kdim]  row-major bf16 (i.e. B stored K-major; stride strideB)
//   C  : row-major [M, N] (or transposed [N, M] if TRANS_OUT), type OutT
// Block = 128 threads (4 waves). Block tile = 128x64. Wave tile = 32x64.
// K loop double-buffered in registers: loads for step k+32 are in flight
// while 8 WMMAs consume step k.
// Requires: M % 128 == 0, N % 64 == 0, Kdim % 64 == 0.
// ---------------------------------------------------------------------------
template <typename OutT, bool TRANS_OUT>
__global__ __launch_bounds__(128)
void k_gemm_bf16(const bf16_t* __restrict__ A, long strideA, int lda,
                 const bf16_t* __restrict__ Bt, long strideB, int ldb,
                 OutT* __restrict__ C, long strideC, int ldc,
                 const float* __restrict__ bias, float alpha, int Kdim) {
  const int lane = threadIdx.x & 31;
  const int wave = threadIdx.x >> 5;   // 0..3
  const int half = lane >> 4;          // 0 or 1
  const int l16  = lane & 15;

  const int m0 = blockIdx.y * 128 + wave * 32;
  const int n0 = blockIdx.x * 64;
  const int b  = blockIdx.z;

  const bf16_t* Ab = A  + (size_t)b * strideA;
  const bf16_t* Bb = Bt + (size_t)b * strideB;

  const bf16_t* arow0 = Ab + (size_t)(m0 + l16) * lda;
  const bf16_t* arow1 = Ab + (size_t)(m0 + 16 + l16) * lda;
  const bf16_t* brow  = Bb + (size_t)(n0 + l16) * ldb;

  v8f acc[2][4] = {};

  Frags f0, f1;
  load_frags(f0, arow0, arow1, brow, ldb, 0, half);

  for (int k0 = 0; k0 < Kdim; k0 += 64) {
    load_frags(f1, arow0, arow1, brow, ldb, k0 + 32, half);
    do_mma(acc, f0);
    if (k0 + 64 < Kdim)
      load_frags(f0, arow0, arow1, brow, ldb, k0 + 64, half);
    do_mma(acc, f1);
  }

  OutT* Cb = C + (size_t)b * strideC;
#pragma unroll
  for (int i = 0; i < 2; ++i) {
#pragma unroll
    for (int j = 0; j < 4; ++j) {
      const int n = n0 + 16 * j + l16;
      const float bv = bias ? bias[n] : 0.0f;
#pragma unroll
      for (int r = 0; r < 8; ++r) {
        const int m = m0 + 16 * i + r + 8 * half;
        const float v = acc[i][j][r] * alpha + bv;
        if (TRANS_OUT) Cb[(size_t)n * ldc + m] = (OutT)v;
        else           Cb[(size_t)m * ldc + n] = (OutT)v;
      }
    }
  }
}

// ---------------------------------------------------------------------------
// In-place row softmax over bf16 rows of length 2048. One block (256 thr) per
// row; each thread owns 8 strided elements; shared-memory tree reductions.
// ---------------------------------------------------------------------------
__global__ __launch_bounds__(256)
void k_softmax_rows_bf16(bf16_t* __restrict__ sp, int rowlen) {
  __shared__ float red[256];
  const int tid = threadIdx.x;
  bf16_t* row = sp + (size_t)blockIdx.x * rowlen;

  float v[8];
#pragma unroll
  for (int i = 0; i < 8; ++i) v[i] = (float)row[tid + 256 * i];

  float m = v[0];
#pragma unroll
  for (int i = 1; i < 8; ++i) m = fmaxf(m, v[i]);
  red[tid] = m;
  __syncthreads();
  for (int s = 128; s > 0; s >>= 1) {
    if (tid < s) red[tid] = fmaxf(red[tid], red[tid + s]);
    __syncthreads();
  }
  m = red[0];
  __syncthreads();

  float sum = 0.0f;
#pragma unroll
  for (int i = 0; i < 8; ++i) { v[i] = __expf(v[i] - m); sum += v[i]; }
  red[tid] = sum;
  __syncthreads();
  for (int s = 128; s > 0; s >>= 1) {
    if (tid < s) red[tid] += red[tid + s];
    __syncthreads();
  }
  const float inv = 1.0f / red[0];
#pragma unroll
  for (int i = 0; i < 8; ++i) row[tid + 256 * i] = (bf16_t)(v[i] * inv);
}

// ---------------------------------------------------------------------------
extern "C" void kernel_launch(void* const* d_in, const int* in_sizes, int n_in,
                              void* d_out, int out_size, void* d_ws, size_t ws_size,
                              hipStream_t stream) {
  (void)in_sizes; (void)n_in; (void)out_size; (void)ws_size;

  const float* X  = (const float*)d_in[0];
  const float* Wq = (const float*)d_in[1];
  const float* bq = (const float*)d_in[2];
  const float* Wk = (const float*)d_in[3];
  const float* bk = (const float*)d_in[4];
  const float* Wv = (const float*)d_in[5];
  const float* bv = (const float*)d_in[6];
  const float* Wo = (const float*)d_in[7];
  const float* bo = (const float*)d_in[8];
  float* out = (float*)d_out;

  const int B = 4, S = 2048, D = 1024, M = B * S;

  uint8_t* ws = (uint8_t*)d_ws;
  size_t off = 0;
  auto take = [&](size_t bytes) -> uint8_t* {
    uint8_t* p = ws + off;
    off += (bytes + 255) & ~(size_t)255;
    return p;
  };
  bf16_t* xbf = (bf16_t*)take((size_t)M * D * 2);       // bf16 X; reused for O
  bf16_t* wtq = (bf16_t*)take((size_t)D * D * 2);       // Wq^T bf16 [N,K]
  bf16_t* wtk = (bf16_t*)take((size_t)D * D * 2);
  bf16_t* wtv = (bf16_t*)take((size_t)D * D * 2);
  bf16_t* wto = (bf16_t*)take((size_t)D * D * 2);
  bf16_t* qbf = (bf16_t*)take((size_t)M * D * 2);       // Q bf16 [M,D]
  bf16_t* kbf = (bf16_t*)take((size_t)M * D * 2);       // K bf16 [M,D]
  bf16_t* vt  = (bf16_t*)take((size_t)M * D * 2);       // V^T bf16 [B][D][S]
  bf16_t* sp  = (bf16_t*)take((size_t)B * S * S * 2);   // scores / probs bf16

  const dim3 blk(128);

  // 1) X -> bf16
  k_f32_to_bf16x4<<<(M * D) / (4 * 256), 256, 0, stream>>>(X, xbf);

  // 2) Weights -> transposed bf16 (K-major for WMMA B operand)
  k_transpose_to_bf16<<<(D * D) / 256, 256, 0, stream>>>(Wq, wtq, D, D);
  k_transpose_to_bf16<<<(D * D) / 256, 256, 0, stream>>>(Wk, wtk, D, D);
  k_transpose_to_bf16<<<(D * D) / 256, 256, 0, stream>>>(Wv, wtv, D, D);
  k_transpose_to_bf16<<<(D * D) / 256, 256, 0, stream>>>(Wo, wto, D, D);

  // 3) Q = X @ Wq + bq   (bf16 out, row-major)
  k_gemm_bf16<bf16_t, false><<<dim3(D / 64, M / 128, 1), blk, 0, stream>>>(
      xbf, 0, D, wtq, 0, D, qbf, 0, D, bq, 1.0f, D);
  // 4) K = X @ Wk + bk
  k_gemm_bf16<bf16_t, false><<<dim3(D / 64, M / 128, 1), blk, 0, stream>>>(
      xbf, 0, D, wtk, 0, D, kbf, 0, D, bk, 1.0f, D);
  // 5) V^T = (X @ Wv + bv)^T per batch -> vt[b][d][s]
  k_gemm_bf16<bf16_t, true><<<dim3(D / 64, S / 128, B), blk, 0, stream>>>(
      xbf, (long)S * D, D, wtv, 0, D, vt, (long)D * S, S, bv, 1.0f, D);

  // 6) scores = (Q @ K^T) / 32   -> sp bf16 [B][S][S]
  k_gemm_bf16<bf16_t, false><<<dim3(S / 64, S / 128, B), blk, 0, stream>>>(
      qbf, (long)S * D, D, kbf, (long)S * D, D, sp, (long)S * S, S,
      nullptr, 0.03125f, D);

  // 7) in-place row softmax
  k_softmax_rows_bf16<<<B * S, 256, 0, stream>>>(sp, S);

  // 8) O = P @ V : A = sp [S,S], Bt = vt [D rows x S] per batch -> obf [M,D]
  bf16_t* obf = xbf;  // X no longer needed
  k_gemm_bf16<bf16_t, false><<<dim3(D / 64, S / 128, B), blk, 0, stream>>>(
      sp, (long)S * S, S, vt, (long)D * S, S, obf, (long)S * D, D,
      nullptr, 1.0f, S);

  // 9) out = O @ Wo + bo  (f32 out)
  k_gemm_bf16<float, false><<<dim3(D / 64, M / 128, 1), blk, 0, stream>>>(
      obf, 0, D, wto, 0, D, out, 0, D, bo, 1.0f, D);
}
